// ContextAttentionBlock_17343077941476
// MI455X (gfx1250) — matrix operand order repaired and checked by reading, not compiled, not run on hardware
//
#include <hip/hip_runtime.h>

typedef __attribute__((ext_vector_type(16))) __bf16 v16bf;
typedef __attribute__((ext_vector_type(8)))  float  v8f;
typedef __attribute__((ext_vector_type(4)))  unsigned int v4u;
typedef __attribute__((ext_vector_type(2)))  unsigned int v2u;

#define LDB   136       // padded LDS row (bf16 elems): 272B = 68 banks
#define TILE  16384     // 128*128

__device__ __forceinline__ v8f wmma_bf16(v16bf a, v16bf b, v8f c) {
  // D = A(16x32 bf16) * B(32x16 bf16) + C(16x16 f32)
  return __builtin_amdgcn_wmma_f32_16x16x32_bf16(false, a, false, b, (short)0, c,
                                                 false, false);
}

// pack two f32 into a bf16x2 dword (folds to v_cvt_pk_bf16_f32)
__device__ __forceinline__ unsigned pk2(float a, float b) {
  __bf16 x = (__bf16)a, y = (__bf16)b;
  unsigned short lo, hi;
  __builtin_memcpy(&lo, &x, 2);
  __builtin_memcpy(&hi, &y, 2);
  return (unsigned)lo | ((unsigned)hi << 16);
}

__device__ __forceinline__ float fast_sigmoid(float z) {
  return __builtin_amdgcn_rcpf(1.0f + __expf(-z));   // v_exp + v_rcp
}

// A fragment (16x32 bf16) from row-major S[m][k]: two contiguous 16B reads.
__device__ __forceinline__ v16bf ldA16(const __bf16* S, int m, int kt, int g) {
  const __bf16* p = S + m * LDB + kt * 32 + g * 8;      // 16B aligned
  union { v4u q[2]; v16bf v; } u;
  u.q[0] = *(const v4u*)p;
  u.q[1] = *(const v4u*)(p + 16);
  return u.v;
}

// B fragment (32x16 bf16) from TRANSPOSED source: element (k,n) <- ST[n][k].
__device__ __forceinline__ v16bf ldB16(const __bf16* ST, int col, int kt, int g) {
  const __bf16* p = ST + col * LDB + kt * 32 + g * 8;   // 16B aligned
  union { v4u q[2]; v16bf v; } u;
  u.q[0] = *(const v4u*)p;
  u.q[1] = *(const v4u*)(p + 16);
  return u.v;
}

// D = x @ W + b ; store row-major and/or transposed (compile-time selected).
// Next tile's B fragments are prefetched before the current tile's WMMAs.
template<bool ROW, bool TR>
__device__ __forceinline__ void conv_gemm(const __bf16* Xs, const __bf16* WT,
                                          const float* __restrict__ bias,
                                          __bf16* Drow, __bf16* Dtr,
                                          int m, int g, int ln, int row0) {
  float bb[8];
#pragma unroll
  for (int nt = 0; nt < 8; ++nt) bb[nt] = bias[nt * 16 + ln];  // one wait, early
  v16bf a0 = ldA16(Xs, m, 0, g), a1 = ldA16(Xs, m, 1, g);
  v16bf a2 = ldA16(Xs, m, 2, g), a3 = ldA16(Xs, m, 3, g);
  v16bf b0 = ldB16(WT, ln, 0, g), b1 = ldB16(WT, ln, 1, g);
  v16bf b2 = ldB16(WT, ln, 2, g), b3 = ldB16(WT, ln, 3, g);
  __bf16* drow = ROW ? (Drow + row0 * LDB + ln) : nullptr;
  __bf16* dtr  = TR  ? (Dtr + ln * LDB + row0)  : nullptr;
#pragma unroll
  for (int nt = 0; nt < 8; ++nt) {
    v16bf n0 = b0, n1 = b1, n2 = b2, n3 = b3;
    if (nt < 7) {                       // prefetch next tile's B fragments
      int nc = (nt + 1) * 16 + ln;
      n0 = ldB16(WT, nc, 0, g); n1 = ldB16(WT, nc, 1, g);
      n2 = ldB16(WT, nc, 2, g); n3 = ldB16(WT, nc, 3, g);
    }
    v8f c = {};
    c = wmma_bf16(a0, b0, c);
    c = wmma_bf16(a1, b1, c);
    c = wmma_bf16(a2, b2, c);
    c = wmma_bf16(a3, b3, c);
    if (ROW) {
#pragma unroll
      for (int r = 0; r < 8; ++r)
        drow[r * LDB + nt * 16] = (__bf16)(c[r] + bb[nt]);   // const offsets
    }
    if (TR) {                            // 8 consecutive rows -> one b128 store
      v4u pk;
      pk.x = pk2(c[0] + bb[nt], c[1] + bb[nt]);
      pk.y = pk2(c[2] + bb[nt], c[3] + bb[nt]);
      pk.z = pk2(c[4] + bb[nt], c[5] + bb[nt]);
      pk.w = pk2(c[6] + bb[nt], c[7] + bb[nt]);
      *(v4u*)&dtr[nt * 16 * LDB] = pk;                        // const offsets
    }
    b0 = n0; b1 = n1; b2 = n2; b3 = n3;
  }
}

// Rs = sigmoid(Asrc @ BTsrc^T) * x
__device__ __forceinline__ void attn_gemm(const __bf16* Xs, const __bf16* Asrc,
                                          const __bf16* BTsrc, __bf16* Rs,
                                          int m, int g, int ln, int row0) {
  v16bf a0 = ldA16(Asrc, m, 0, g), a1 = ldA16(Asrc, m, 1, g);
  v16bf a2 = ldA16(Asrc, m, 2, g), a3 = ldA16(Asrc, m, 3, g);
  v16bf b0 = ldB16(BTsrc, ln, 0, g), b1 = ldB16(BTsrc, ln, 1, g);
  v16bf b2 = ldB16(BTsrc, ln, 2, g), b3 = ldB16(BTsrc, ln, 3, g);
  const __bf16* xrow = Xs + row0 * LDB + ln;
  __bf16* rrow = Rs + row0 * LDB + ln;
#pragma unroll
  for (int nt = 0; nt < 8; ++nt) {
    v16bf n0 = b0, n1 = b1, n2 = b2, n3 = b3;
    if (nt < 7) {                       // prefetch; overlaps sigmoid epilogue
      int nc = (nt + 1) * 16 + ln;
      n0 = ldB16(BTsrc, nc, 0, g); n1 = ldB16(BTsrc, nc, 1, g);
      n2 = ldB16(BTsrc, nc, 2, g); n3 = ldB16(BTsrc, nc, 3, g);
    }
    v8f c = {};
    c = wmma_bf16(a0, b0, c);
    c = wmma_bf16(a1, b1, c);
    c = wmma_bf16(a2, b2, c);
    c = wmma_bf16(a3, b3, c);
#pragma unroll
    for (int r = 0; r < 8; ++r) {
      float s = fast_sigmoid(c[r]);
      rrow[r * LDB + nt * 16] =
          (__bf16)(s * (float)xrow[r * LDB + nt * 16]);       // const offsets
    }
    b0 = n0; b1 = n1; b2 = n2; b3 = n3;
  }
}

// facc += Asrc @ W  (one 128-K slice of the final 384-K GEMM)
__device__ __forceinline__ void fin_gemm(v8f (&facc)[8], const __bf16* Asrc,
                                         const __bf16* WT, int m, int g, int ln) {
  v16bf a0 = ldA16(Asrc, m, 0, g), a1 = ldA16(Asrc, m, 1, g);
  v16bf a2 = ldA16(Asrc, m, 2, g), a3 = ldA16(Asrc, m, 3, g);
  v16bf b0 = ldB16(WT, ln, 0, g), b1 = ldB16(WT, ln, 1, g);
  v16bf b2 = ldB16(WT, ln, 2, g), b3 = ldB16(WT, ln, 3, g);
#pragma unroll
  for (int nt = 0; nt < 8; ++nt) {
    v16bf n0 = b0, n1 = b1, n2 = b2, n3 = b3;
    if (nt < 7) {
      int nc = (nt + 1) * 16 + ln;
      n0 = ldB16(WT, nc, 0, g); n1 = ldB16(WT, nc, 1, g);
      n2 = ldB16(WT, nc, 2, g); n3 = ldB16(WT, nc, 3, g);
    }
    facc[nt] = wmma_bf16(a0, b0, facc[nt]);
    facc[nt] = wmma_bf16(a1, b1, facc[nt]);
    facc[nt] = wmma_bf16(a2, b2, facc[nt]);
    facc[nt] = wmma_bf16(a3, b3, facc[nt]);
    b0 = n0; b1 = n1; b2 = n2; b3 = n3;
  }
}

__global__ void __launch_bounds__(256, 1)
cab_kernel(const float* __restrict__ x,
           const float* __restrict__ w_theta, const float* __restrict__ b_theta,
           const float* __restrict__ w_phi,   const float* __restrict__ b_phi,
           const float* __restrict__ w_g,     const float* __restrict__ b_g,
           const float* __restrict__ w_sc,    const float* __restrict__ b_sc,
           const float* __restrict__ w_out,   const float* __restrict__ b_out,
           float* __restrict__ out)
{
  extern __shared__ __bf16 smem[];
  __bf16* Xs = smem;              // x strip, row-major (A for all convs)
  __bf16* Ts = Xs + 128 * LDB;    // theta (row-major) / f_g^T / short_cut
  __bf16* Ps = Ts + 128 * LDB;    // phi row-major (A of horizontal)
  __bf16* PT = Ps + 128 * LDB;    // phi transposed (B of vertical)
  __bf16* Rs = PT + 128 * LDB;    // horizontal / vertical result (row-major)
  __bf16* W0 = Rs + 128 * LDB;    // weight^T ping
  __bf16* W1 = W0 + 128 * LDB;    // weight^T pong

  const int tid  = threadIdx.x;
  const int lane = tid & 31;
  const int wv   = tid >> 5;        // wave id == M-tile
  const int g    = lane >> 4;
  const int ln   = lane & 15;
  const int m    = wv * 16 + ln;    // A-row owned by this lane
  const int row0 = wv * 16 + 8 * g; // first C/D row owned by this lane

  const float* xin  = x   + (size_t)blockIdx.x * TILE;
  float*       oput = out + (size_t)blockIdx.x * TILE;

  v8f facc[8] = {};                 // final-GEMM accumulators

  auto stage_x = [&]() {            // f32 global -> bf16 row-major LDS
    for (int i = tid; i < 4096; i += 256) {
      float4 q = ((const float4*)xin)[i];
      v2u pk;
      pk.x = pk2(q.x, q.y);
      pk.y = pk2(q.z, q.w);
      *(v2u*)&Xs[(i >> 5) * LDB + (i & 31) * 4] = pk;    // 8B aligned
    }
  };

  // stage a [128,128] f32 weight TRANSPOSED: WT[n][k] = w[k][n]
  auto stage_wT = [&](const float* __restrict__ src, __bf16* WT) {
    for (int i = tid; i < 2048; i += 256) {
      int c = (i >> 5) * 2;       // source row pair (K)
      int d = (i & 31) * 4;       // source col group (N)
      float4 q0 = *(const float4*)(src + c * 128 + d);
      float4 q1 = *(const float4*)(src + (c + 1) * 128 + d);
      *(unsigned*)&WT[(d + 0) * LDB + c] = pk2(q0.x, q1.x);
      *(unsigned*)&WT[(d + 1) * LDB + c] = pk2(q0.y, q1.y);
      *(unsigned*)&WT[(d + 2) * LDB + c] = pk2(q0.z, q1.z);
      *(unsigned*)&WT[(d + 3) * LDB + c] = pk2(q0.w, q1.w);
    }
  };

  // ---- 10-phase pipeline, every weight stage overlapped with a GEMM ----
  stage_x();
  stage_wT(w_theta, W0);
  stage_wT(w_phi, W1);
  __syncthreads();
  conv_gemm<true, false>(Xs, W0, b_theta, Ts, nullptr, m, g, ln, row0);
  __syncthreads();
  conv_gemm<true, true>(Xs, W1, b_phi, Ps, PT, m, g, ln, row0);
  stage_wT(w_g, W0);                  // overlap: W0 free since last phase
  __syncthreads();
  attn_gemm(Xs, Ps, Ts, Rs, m, g, ln, row0);   // horizontal -> Rs
  stage_wT(w_out, W1);                // Wout[0:128]^T
  __syncthreads();
  fin_gemm(facc, Rs, W1, m, g, ln);   // out += horizontal @ Wout[0:128]
  __syncthreads();
  conv_gemm<false, true>(Xs, W0, b_g, nullptr, Ts, m, g, ln, row0); // f_g^T
  stage_wT(w_out + 2 * TILE, W1);     // Wout[256:384]^T
  __syncthreads();
  attn_gemm(Xs, Ts, PT, Rs, m, g, ln, row0);   // vertical -> Rs
  stage_wT(w_sc, W0);
  __syncthreads();
  fin_gemm(facc, Rs, W1, m, g, ln);   // out += vertical @ Wout[256:384]
  __syncthreads();
  conv_gemm<true, false>(Xs, W0, b_sc, Ts, nullptr, m, g, ln, row0); // sc
  stage_wT(w_out + TILE, W1);         // Wout[128:256]^T
  __syncthreads();
  fin_gemm(facc, Ts, W1, m, g, ln);   // out += short_cut @ Wout[128:256]

  // epilogue: add b_out (preloaded), store fp32
  float bb[8];
#pragma unroll
  for (int nt = 0; nt < 8; ++nt) bb[nt] = b_out[nt * 16 + ln];
  float* orow = oput + row0 * 128 + ln;
#pragma unroll
  for (int nt = 0; nt < 8; ++nt) {
#pragma unroll
    for (int r = 0; r < 8; ++r)
      orow[r * 128 + nt * 16] = facc[nt][r] + bb[nt];
  }
}

extern "C" void kernel_launch(void* const* d_in, const int* in_sizes, int n_in,
                              void* d_out, int out_size, void* d_ws, size_t ws_size,
                              hipStream_t stream) {
  const float* x    = (const float*)d_in[0];
  const float* wt   = (const float*)d_in[1];
  const float* bt   = (const float*)d_in[2];
  const float* wp   = (const float*)d_in[3];
  const float* bp   = (const float*)d_in[4];
  const float* wg   = (const float*)d_in[5];
  const float* bg   = (const float*)d_in[6];
  const float* wsc  = (const float*)d_in[7];
  const float* bsc  = (const float*)d_in[8];
  const float* wo   = (const float*)d_in[9];
  const float* bo   = (const float*)d_in[10];
  float* out = (float*)d_out;

  int nstrips = in_sizes[0] / TILE;          // B*H = 2048
  size_t shmem = (size_t)7 * 128 * LDB * 2;  // 243,712 B of 320 KB WGP LDS
  cab_kernel<<<nstrips, 256, shmem, stream>>>(x, wt, bt, wp, bp, wg, bg,
                                              wsc, bsc, wo, bo, out);
}
